// StructureModule_90323162235229
// MI455X (gfx1250) — compile-verified, matrix operand order checked by reading, not run on hardware
//
#include <hip/hip_runtime.h>
#include <hip/hip_bf16.h>
#include <math.h>

#define II 384
#define CC 128
#define HH 8
#define QPP 4
#define VPP 8
#define NLL 8
#define CATD 2304          // H*C + H*C + 3*H*VP + H*VP
#define WC_CONST 0.23570226039551584f   // sqrt(2/(9*QP))
#define WL_CONST 0.57735026918962576f   // sqrt(1/3)

typedef __attribute__((ext_vector_type(16))) _Float16 v16h;
typedef __attribute__((ext_vector_type(8)))  float    v8f;

// ------------------------------------------------------------------
// Generic WMMA GEMM (REQUIRES K % 32 == 0):
//   Out[b,m,n] = act( alpha * sum_k A[b,m,k]*B[b,k,n] + biasN[n]
//                     + addScale*addF[b,m,n] )
// A addr  : A + b*sAb + m*sAm + k              (optional ReLU on load)
// B addr  : bT==0 : B + b*sBb + k*sB + n
//           bT==1 : B + b*sBb + n*sB + k       (i.e. B given as N x K)
// Out addr: Out + b*sOb + m*sOm + n*sOn
// One wave (32 lanes) computes one 16x16 tile via v_wmma_f32_16x16x32_f16.
// Edge tiles: indices are CLAMPED (loads stay unconditional / branchless),
// out-of-range contributions are zeroed by a 0/1 multiplier (v_cndmask on
// data, not exec), so the K-loop carries no divergent branches.
// Contiguous-K operands use float4 (global_load_b128); strided-K B uses
// 16 unconditional b32 loads whose offsets are hoisted out of the K-loop
// (no per-iteration 64-bit multiplies, just two pointer bumps).
// ------------------------------------------------------------------
__global__ __launch_bounds__(32)
void wmma_gemm_kernel(const float* __restrict__ A, long sAm, long sAb, int preReluA,
                      const float* __restrict__ B, long sB, long sBb, int bT,
                      float* __restrict__ Out, long sOm, long sOn, long sOb,
                      const float* __restrict__ biasN,
                      const float* __restrict__ addF, long sAddm, long sAddb, float addScale,
                      float alpha, int act, int M, int N, int K)
{
    const int lane  = threadIdx.x;
    const int half  = lane >> 4;        // 0: lanes 0-15, 1: lanes 16-31
    const int idx16 = lane & 15;
    const int tm = blockIdx.y * 16;
    const int tn = blockIdx.x * 16;
    const int b  = blockIdx.z;

    const float* Ab = A + (long)b * sAb;
    const float* Bb = B + (long)b * sBb;

    const int mWant = tm + idx16;
    const int nWant = tn + idx16;
    const int mA = (mWant < M) ? mWant : (M - 1);      // clamped, always valid
    const int nB = (nWant < N) ? nWant : (N - 1);
    const float mMul = (mWant < M) ? 1.0f : 0.0f;
    const float nMul = (nWant < N) ? 1.0f : 0.0f;

    // Row base for this lane's A fragment; the fragment's K offsets are
    // {k0+0..7} and {k0+16..23} relative to this (half adds +8).
    const float* Arow = Ab + (long)mA * sAm + (half ? 8 : 0);

    v8f acc = {};

    if (bT) {
        // B is N x K (k-contiguous) -> same vector path as A.
        const float* Brow = Bb + (long)nB * sB + (half ? 8 : 0);
        for (int k0 = 0; k0 < K; k0 += 32) {
            float4 a0 = *(const float4*)(Arow + k0);
            float4 a1 = *(const float4*)(Arow + k0 + 4);
            float4 a2 = *(const float4*)(Arow + k0 + 16);
            float4 a3 = *(const float4*)(Arow + k0 + 20);
            float4 b0 = *(const float4*)(Brow + k0);
            float4 b1 = *(const float4*)(Brow + k0 + 4);
            float4 b2 = *(const float4*)(Brow + k0 + 16);
            float4 b3 = *(const float4*)(Brow + k0 + 20);
            __builtin_prefetch(Arow + k0 + 32, 0, 3);
            __builtin_prefetch(Brow + k0 + 32, 0, 3);
            float av[16] = { a0.x,a0.y,a0.z,a0.w, a1.x,a1.y,a1.z,a1.w,
                             a2.x,a2.y,a2.z,a2.w, a3.x,a3.y,a3.z,a3.w };
            float bv[16] = { b0.x,b0.y,b0.z,b0.w, b1.x,b1.y,b1.z,b1.w,
                             b2.x,b2.y,b2.z,b2.w, b3.x,b3.y,b3.z,b3.w };
            v16h af, bf;
#pragma unroll
            for (int e = 0; e < 16; ++e) {
                float a = av[e];
                if (preReluA) a = fmaxf(a, 0.0f);
                af[e] = (_Float16)(a * mMul);
                bf[e] = (_Float16)(bv[e] * nMul);
            }
            acc = __builtin_amdgcn_wmma_f32_16x16x32_f16(false, af, false, bf,
                                                         (short)0, acc, false, false);
        }
    } else {
        // B is K x N: k-strided on the B side. Hoist the 8 row offsets out of
        // the loop; advance two group base pointers (K+0..7 / K+16..23) by
        // 32*sB per step -> inner loop has no address multiplies.
        const float* Bp1 = Bb + nB + (long)(half ? 8 : 0) * sB;  // K = 0..7 (+8 hi)
        const float* Bp2 = Bp1 + 16 * sB;                        // K = 16..23 (+8 hi)
        long ofs[8];
#pragma unroll
        for (int e = 0; e < 8; ++e) ofs[e] = (long)e * sB;
        const long bStep = 32 * sB;
        for (int k0 = 0; k0 < K; k0 += 32) {
            float4 a0 = *(const float4*)(Arow + k0);
            float4 a1 = *(const float4*)(Arow + k0 + 4);
            float4 a2 = *(const float4*)(Arow + k0 + 16);
            float4 a3 = *(const float4*)(Arow + k0 + 20);
            __builtin_prefetch(Arow + k0 + 32, 0, 3);
            float av[16] = { a0.x,a0.y,a0.z,a0.w, a1.x,a1.y,a1.z,a1.w,
                             a2.x,a2.y,a2.z,a2.w, a3.x,a3.y,a3.z,a3.w };
            v16h af, bf;
#pragma unroll
            for (int e = 0; e < 8; ++e) {
                bf[e]     = (_Float16)(Bp1[ofs[e]] * nMul);
                bf[e + 8] = (_Float16)(Bp2[ofs[e]] * nMul);
            }
#pragma unroll
            for (int e = 0; e < 16; ++e) {
                float a = av[e];
                if (preReluA) a = fmaxf(a, 0.0f);
                af[e] = (_Float16)(a * mMul);
            }
            acc = __builtin_amdgcn_wmma_f32_16x16x32_f16(false, af, false, bf,
                                                         (short)0, acc, false, false);
            Bp1 += bStep;
            Bp2 += bStep;
        }
    }

    float* Ob = Out + (long)b * sOb;
#pragma unroll
    for (int r = 0; r < 8; ++r) {
        int m = tm + r + (half ? 8 : 0);    // C/D layout: vgpr r -> M=r (+8 for hi lanes)
        int n = tn + idx16;
        if (m < M && n < N) {
            float v = alpha * acc[r];
            if (biasN) v += biasN[n];
            if (addF)  v += addScale * addF[(long)b * sAddb + (long)m * sAddm + n];
            if (act == 1) v = fmaxf(v, 0.0f);
            Ob[(long)m * sOm + (long)n * sOn] = v;
        }
    }
}

// ---------------- LayerNorm: one wave per row ----------------
__global__ void ln_kernel(const float* __restrict__ x, const float* __restrict__ g,
                          const float* __restrict__ bta, float* __restrict__ y,
                          long rows, int cols)
{
    long row  = (long)blockIdx.x * (blockDim.x >> 5) + (threadIdx.x >> 5);
    int  lane = threadIdx.x & 31;
    if (row >= rows) return;
    const float* xr = x + row * cols;
    float s = 0.f, sq = 0.f;
    for (int c = lane; c < cols; c += 32) { float v = xr[c]; s += v; sq += v * v; }
    for (int o = 16; o; o >>= 1) { s += __shfl_xor(s, o, 32); sq += __shfl_xor(sq, o, 32); }
    float mean = s / cols;
    float var  = sq / cols - mean * mean;
    float inv  = rsqrtf(var + 1e-5f);
    float* yr = y + row * cols;
    for (int c = lane; c < cols; c += 32)
        yr[c] = (xr[c] - mean) * inv * g[c] + bta[c];
}

// ---------------- frames: R=I, t=0 ----------------
__global__ void init_frames_kernel(float* __restrict__ R, float* __restrict__ t)
{
    int i = blockIdx.x * blockDim.x + threadIdx.x;
    if (i >= II) return;
    float* Ri = R + i * 9;
    Ri[0] = 1.f; Ri[1] = 0.f; Ri[2] = 0.f;
    Ri[3] = 0.f; Ri[4] = 1.f; Ri[5] = 0.f;
    Ri[6] = 0.f; Ri[7] = 0.f; Ri[8] = 1.f;
    t[i*3] = 0.f; t[i*3+1] = 0.f; t[i*3+2] = 0.f;
}

// ---------------- points to global frame: out = R x + t ----------------
__global__ void to_global_kernel(const float* __restrict__ loc, const float* __restrict__ R,
                                 const float* __restrict__ t, float* __restrict__ outp, int P)
{
    long idx = (long)blockIdx.x * blockDim.x + threadIdx.x;   // i*P + p
    if (idx >= (long)II * P) return;
    int i = (int)(idx / P);
    const float* x  = loc + idx * 3;
    const float* Rm = R + i * 9;
    const float* tv = t + i * 3;
    float x0 = x[0], x1 = x[1], x2 = x[2];
    outp[idx*3+0] = Rm[0]*x0 + Rm[1]*x1 + Rm[2]*x2 + tv[0];
    outp[idx*3+1] = Rm[3]*x0 + Rm[4]*x1 + Rm[5]*x2 + tv[1];
    outp[idx*3+2] = Rm[6]*x0 + Rm[7]*x1 + Rm[8]*x2 + tv[2];
}

// ---------------- a = (a + hw[h]*pt) * W_L ----------------
__global__ void pt_bias_kernel(float* __restrict__ a, const float* __restrict__ qpg,
                               const float* __restrict__ kpg, const float* __restrict__ gamma)
{
    long idx = (long)blockIdx.x * blockDim.x + threadIdx.x;    // [h][i][j]
    const long tot = (long)HH * II * II;
    if (idx >= tot) return;
    int j = (int)(idx % II);
    long r = idx / II;
    int i = (int)(r % II);
    int h = (int)(r / II);
    const float* qp = qpg + ((long)i * HH + h) * (QPP * 3);
    const float* kp = kpg + ((long)j * HH + h) * (QPP * 3);
    float pt = 0.f;
#pragma unroll
    for (int p = 0; p < QPP; ++p)
#pragma unroll
        for (int d = 0; d < 3; ++d) {
            float df = qp[p*3+d] - kp[p*3+d];
            pt += df * df;
        }
    float g  = gamma[h];
    float sp = (g > 20.f) ? g : log1pf(__expf(g));   // softplus
    float hw = sp * WC_CONST * (-0.5f);
    a[idx] = (a[idx] + hw * pt) * WL_CONST;
}

// ---------------- softmax over query axis i for each (h, j) ----------------
__global__ void softmax_q_kernel(float* __restrict__ a)
{
    int wid  = (blockIdx.x * blockDim.x + threadIdx.x) >> 5;   // h*II + j
    int lane = threadIdx.x & 31;
    if (wid >= HH * II) return;
    int h = wid / II, j = wid % II;
    float* base = a + (long)h * II * II + j;
    float mx = -3.4e38f;
    for (int i = lane; i < II; i += 32) mx = fmaxf(mx, base[(long)i * II]);
    for (int o = 16; o; o >>= 1) mx = fmaxf(mx, __shfl_xor(mx, o, 32));
    float sum = 0.f;
    for (int i = lane; i < II; i += 32) {
        float e = __expf(base[(long)i * II] - mx);
        base[(long)i * II] = e;
        sum += e;
    }
    for (int o = 16; o; o >>= 1) sum += __shfl_xor(sum, o, 32);
    float inv = 1.0f / sum;
    for (int i = lane; i < II; i += 32) base[(long)i * II] *= inv;
}

// ---------------- asum[h,i] = sum_j a[h,i,j] ----------------
__global__ void rowsum_kernel(const float* __restrict__ a, float* __restrict__ asum)
{
    int wid  = (blockIdx.x * blockDim.x + threadIdx.x) >> 5;   // h*II + i
    int lane = threadIdx.x & 31;
    if (wid >= HH * II) return;
    const float* row = a + (long)wid * II;
    float s = 0.f;
    for (int j = lane; j < II; j += 32) s += row[j];
    for (int o = 16; o; o >>= 1) s += __shfl_xor(s, o, 32);
    if (lane == 0) asum[wid] = s;
}

// ---------------- o[i,h,c] = asum[h,i] * v[i,h,c]  -> cat[:,0:1024] ----------------
__global__ void o_scale_kernel(const float* __restrict__ asum, const float* __restrict__ v,
                               float* __restrict__ cat)
{
    long idx = (long)blockIdx.x * blockDim.x + threadIdx.x;    // i*(H*C)+h*C+c
    if (idx >= (long)II * HH * CC) return;
    int hc = (int)(idx % (HH * CC));
    int i  = (int)(idx / (HH * CC));
    int h  = hc / CC;
    cat[(long)i * CATD + hc] = asum[h * II + i] * v[idx];
}

// ---------------- op_norm: cat[i,2240+hp] = max(||op||,1e-6) ----------------
__global__ void opnorm_kernel(float* __restrict__ cat)
{
    long idx = (long)blockIdx.x * blockDim.x + threadIdx.x;    // i*(H*VP)+hp
    if (idx >= (long)II * HH * VPP) return;
    int hp = (int)(idx % (HH * VPP));
    int i  = (int)(idx / (HH * VPP));
    const float* p = cat + (long)i * CATD + 2 * HH * CC + hp * 3;
    float n = sqrtf(p[0]*p[0] + p[1]*p[1] + p[2]*p[2]);
    cat[(long)i * CATD + 2 * HH * CC + 3 * HH * VPP + hp] = fmaxf(n, 1e-6f);
}

// ---------------- backbone update + rigid compose (per residue) ----------------
__global__ void backbone_kernel(const float* __restrict__ s,
                                const float* __restrict__ Wb, const float* __restrict__ bb,
                                const float* __restrict__ Wc, const float* __restrict__ bc,
                                const float* __restrict__ Wd, const float* __restrict__ bd,
                                const float* __restrict__ Wt, const float* __restrict__ bt,
                                float* __restrict__ R, float* __restrict__ t)
{
    int i = blockIdx.x * blockDim.x + threadIdx.x;
    if (i >= II) return;
    const float* sr = s + (long)i * CC;
    float qb = bb[0], qc = bc[0], qd = bd[0];
    float t0 = bt[0], t1 = bt[1], t2 = bt[2];
    for (int c = 0; c < CC; ++c) {
        float sv = sr[c];
        qb += sv * Wb[c];
        qc += sv * Wc[c];
        qd += sv * Wd[c];
        t0 += sv * Wt[c*3+0];
        t1 += sv * Wt[c*3+1];
        t2 += sv * Wt[c*3+2];
    }
    float inv = rsqrtf(1.f + qb*qb + qc*qc + qd*qd);
    float w = inv, x = qb*inv, y = qc*inv, z = qd*inv;
    float Ru[9] = {
        1-2*(y*y+z*z), 2*(x*y-w*z),   2*(x*z+w*y),
        2*(x*y+w*z),   1-2*(x*x+z*z), 2*(y*z-w*x),
        2*(x*z-w*y),   2*(y*z+w*x),   1-2*(x*x+y*y)
    };
    float* Ri = R + i * 9;
    float* ti = t + i * 3;
    float Ro[9]; for (int k = 0; k < 9; ++k) Ro[k] = Ri[k];
    float to_[3] = { ti[0], ti[1], ti[2] };
    float tu[3] = { t0, t1, t2 };
    for (int m = 0; m < 3; ++m)
        ti[m] = Ro[m*3]*tu[0] + Ro[m*3+1]*tu[1] + Ro[m*3+2]*tu[2] + to_[m];
    for (int m = 0; m < 3; ++m)
        for (int p = 0; p < 3; ++p) {
            float acc = 0.f;
            for (int n = 0; n < 3; ++n) acc += Ro[m*3+n] * Ru[n*3+p];
            Ri[m*3+p] = acc;
        }
}

// ---------------- angle head finalize: unnorm + normalized ----------------
__global__ void angle_final_kernel(const float* __restrict__ a14,
                                   float* __restrict__ out_unnorm,
                                   float* __restrict__ out_norm)
{
    int idx = blockIdx.x * blockDim.x + threadIdx.x;   // i*7 + k
    if (idx >= II * 7) return;
    float x = a14[idx*2], y = a14[idx*2+1];
    float inv = 1.0f / sqrtf(x*x + y*y);
    out_unnorm[idx*2]   = x;
    out_unnorm[idx*2+1] = y;
    out_norm[idx*2]     = x * inv;
    out_norm[idx*2+1]   = y * inv;
}

__global__ void copy_kernel(const float* __restrict__ src, float* __restrict__ dst, long n)
{
    long idx = (long)blockIdx.x * blockDim.x + threadIdx.x;
    if (idx < n) dst[idx] = src[idx];
}

// ==================================================================
// Host side
// ==================================================================
enum {
    IN_S = 0, IN_Z, IN_AATYPE,
    LN_S_G, LN_S_B, LN_Z_G, LN_Z_B, LIN_IN_W, LIN_IN_B,
    WQ, WK, WV, WBIAS, WQP, WKP, WVP, GAMMA, WOUT_IPA, BOUT_IPA,
    LN_IPA_G, LN_IPA_B, TR_W1, TR_B1, TR_W2, TR_B2, TR_W3, TR_B3,
    TR_LN_G, TR_LN_B, BB_WB, BB_BB, BB_WC, BB_BC, BB_WD, BB_BD, BB_WT, BB_BT,
    ANG_WIN, ANG_BIN, ANG_WINIT, ANG_BINIT, ANG_W1, ANG_B1, ANG_W2, ANG_B2,
    ANG_WOUT, ANG_BOUT
};

static void gemm(hipStream_t st,
                 const float* A, long sAm, long sAb, int relA,
                 const float* B, long sB, long sBb, int bT,
                 float* O, long sOm, long sOn, long sOb,
                 const float* biasN,
                 const float* addF, long sAddm, long sAddb, float addScale,
                 float alpha, int act, int M, int N, int K, int batch)
{
    dim3 grid((N + 15) / 16, (M + 15) / 16, batch);
    wmma_gemm_kernel<<<grid, dim3(32), 0, st>>>(
        A, sAm, sAb, relA, B, sB, sBb, bT, O, sOm, sOn, sOb,
        biasN, addF, sAddm, sAddb, addScale, alpha, act, M, N, K);
}

extern "C" void kernel_launch(void* const* d_in, const int* in_sizes, int n_in,
                              void* d_out, int out_size, void* d_ws, size_t ws_size,
                              hipStream_t stream)
{
    (void)in_sizes; (void)n_in; (void)out_size; (void)ws_size;
    const float* P[48];
    for (int i = 0; i < 47 && i < n_in; ++i) P[i] = (const float*)d_in[i];

    const float* s_init = P[IN_S];
    const float* z_in   = P[IN_Z];

    float* ws = (float*)d_ws;
    size_t off = 0;
    float* z_ln  = ws + off; off += (size_t)II * II * CC;
    float* biasB = ws + off; off += (size_t)HH * II * II;
    float* aB    = ws + off; off += (size_t)HH * II * II;
    float* s0    = ws + off; off += (size_t)II * CC;
    float* sB    = ws + off; off += (size_t)II * CC;
    float* qB    = ws + off; off += (size_t)II * HH * CC;
    float* kB    = ws + off; off += (size_t)II * HH * CC;
    float* vB    = ws + off; off += (size_t)II * HH * CC;
    float* ploc  = ws + off; off += (size_t)II * HH * VPP * 3;   // max local pts
    float* qpg   = ws + off; off += (size_t)II * HH * QPP * 3;
    float* kpg   = ws + off; off += (size_t)II * HH * QPP * 3;
    float* vpg   = ws + off; off += (size_t)II * HH * VPP * 3;
    float* asum  = ws + off; off += (size_t)HH * II;
    float* catB  = ws + off; off += (size_t)II * CATD;
    float* h1B   = ws + off; off += (size_t)II * CC;
    float* h2B   = ws + off; off += (size_t)II * CC;
    float* RB    = ws + off; off += (size_t)II * 9 + 7;
    float* tB    = ws + off; off += (size_t)II * 3 + 13;
    float* angA  = ws + off; off += (size_t)II * CC;
    float* angT  = ws + off; off += (size_t)II * CC;
    float* a14   = ws + off; off += (size_t)II * 14;

    const float inv_sqrt_c = 0.088388347648318447f;   // 1/sqrt(128)

    // ---- setup ----
    ln_kernel<<<(II + 7) / 8, 256, 0, stream>>>(s_init, P[LN_S_G], P[LN_S_B], s0, II, CC);
    ln_kernel<<<((long)II * II + 7) / 8, 256, 0, stream>>>(z_in, P[LN_Z_G], P[LN_Z_B], z_ln, (long)II * II, CC);

    // s = s0 @ lin_in_W + b
    gemm(stream, s0, CC, 0, 0, P[LIN_IN_W], CC, 0, 0, sB, CC, 1, 0,
         P[LIN_IN_B], nullptr, 0, 0, 0.f, 1.f, 0, II, CC, CC, 1);

    // bias[h,i,j] = z_ln @ Wbias (Out written transposed: n-stride = I*I)
    gemm(stream, z_ln, CC, 0, 0, P[WBIAS], HH, 0, 0, biasB, 1, (long)II * II, 0,
         nullptr, nullptr, 0, 0, 0.f, 1.f, 0, II * II, HH, CC, 1);

    init_frames_kernel<<<(II + 255) / 256, 256, 0, stream>>>(RB, tB);

    // ---- structure iterations ----
    for (int it = 0; it < NLL; ++it) {
        // q/k/v projections: (384x128)@(128x1024)
        gemm(stream, sB, CC, 0, 0, P[WQ], HH * CC, 0, 0, qB, HH * CC, 1, 0,
             nullptr, nullptr, 0, 0, 0.f, 1.f, 0, II, HH * CC, CC, 1);
        gemm(stream, sB, CC, 0, 0, P[WK], HH * CC, 0, 0, kB, HH * CC, 1, 0,
             nullptr, nullptr, 0, 0, 0.f, 1.f, 0, II, HH * CC, CC, 1);
        gemm(stream, sB, CC, 0, 0, P[WV], HH * CC, 0, 0, vB, HH * CC, 1, 0,
             nullptr, nullptr, 0, 0, 0.f, 1.f, 0, II, HH * CC, CC, 1);

        // point projections + to-global (R x + t)
        gemm(stream, sB, CC, 0, 0, P[WQP], 3 * QPP * HH, 0, 0, ploc, 3 * QPP * HH, 1, 0,
             nullptr, nullptr, 0, 0, 0.f, 1.f, 0, II, 3 * QPP * HH, CC, 1);
        to_global_kernel<<<((long)II * HH * QPP + 255) / 256, 256, 0, stream>>>(ploc, RB, tB, qpg, HH * QPP);
        gemm(stream, sB, CC, 0, 0, P[WKP], 3 * QPP * HH, 0, 0, ploc, 3 * QPP * HH, 1, 0,
             nullptr, nullptr, 0, 0, 0.f, 1.f, 0, II, 3 * QPP * HH, CC, 1);
        to_global_kernel<<<((long)II * HH * QPP + 255) / 256, 256, 0, stream>>>(ploc, RB, tB, kpg, HH * QPP);
        gemm(stream, sB, CC, 0, 0, P[WVP], 3 * VPP * HH, 0, 0, ploc, 3 * VPP * HH, 1, 0,
             nullptr, nullptr, 0, 0, 0.f, 1.f, 0, II, 3 * VPP * HH, CC, 1);
        to_global_kernel<<<((long)II * HH * VPP + 255) / 256, 256, 0, stream>>>(ploc, RB, tB, vpg, HH * VPP);

        // a[h,i,j] = (q_h @ k_h^T) / sqrt(C) + bias[h,i,j]   (batched over heads)
        gemm(stream, qB, HH * CC, CC, 0, kB, HH * CC, CC, 1,
             aB, II, 1, (long)II * II, nullptr,
             biasB, II, (long)II * II, 1.f, inv_sqrt_c, 0, II, II, CC, HH);

        // a = (a + hw[h]*pt) * W_L ; softmax over query axis i ; row sums over j
        pt_bias_kernel<<<((long)HH * II * II + 255) / 256, 256, 0, stream>>>(aB, qpg, kpg, P[GAMMA]);
        softmax_q_kernel<<<(HH * II + 7) / 8, 256, 0, stream>>>(aB);
        rowsum_kernel<<<(HH * II + 7) / 8, 256, 0, stream>>>(aB, asum);

        // o = sum_j(a) * v  -> cat[:, 0:1024]
        o_scale_kernel<<<((long)II * HH * CC + 255) / 256, 256, 0, stream>>>(asum, vB, catB);

        // o_pair[i,h,c] = sum_j a[h,i,j] z[i,j,c]  (batched over i, M=8) -> cat[:,1024:2048]
        gemm(stream, aB, (long)II * II, II, 0, z_ln, CC, (long)II * CC, 0,
             catB + HH * CC, CC, 1, CATD, nullptr, nullptr, 0, 0, 0.f,
             1.f, 0, HH, CC, II, II);

        // op[i,h,p,m] = sum_j a[h,i,j] vp[j,h,p,m]  (batched over heads) -> cat[:,2048:2240]
        gemm(stream, aB, II, (long)II * II, 0, vpg, HH * VPP * 3, VPP * 3, 0,
             catB + 2 * HH * CC, CATD, 1, VPP * 3, nullptr, nullptr, 0, 0, 0.f,
             1.f, 0, II, VPP * 3, II, HH);

        // op_norm -> cat[:,2240:2304]
        opnorm_kernel<<<((long)II * HH * VPP + 255) / 256, 256, 0, stream>>>(catB);

        // s = s + cat @ Wout_ipa + bout
        gemm(stream, catB, CATD, 0, 0, P[WOUT_IPA], CC, 0, 0, sB, CC, 1, 0,
             P[BOUT_IPA], sB, CC, 0, 1.f, 1.f, 0, II, CC, CATD, 1);
        ln_kernel<<<(II + 7) / 8, 256, 0, stream>>>(sB, P[LN_IPA_G], P[LN_IPA_B], sB, II, CC);

        // transition: 8 layers of relu-relu-residual MLP
        for (int l = 0; l < NLL; ++l) {
            const float* W1 = P[TR_W1] + (long)l * CC * CC;
            const float* W2 = P[TR_W2] + (long)l * CC * CC;
            const float* W3 = P[TR_W3] + (long)l * CC * CC;
            gemm(stream, sB, CC, 0, 0, W1, CC, 0, 0, h1B, CC, 1, 0,
                 P[TR_B1] + l * CC, nullptr, 0, 0, 0.f, 1.f, 1, II, CC, CC, 1);
            gemm(stream, h1B, CC, 0, 0, W2, CC, 0, 0, h2B, CC, 1, 0,
                 P[TR_B2] + l * CC, nullptr, 0, 0, 0.f, 1.f, 1, II, CC, CC, 1);
            gemm(stream, h2B, CC, 0, 0, W3, CC, 0, 0, sB, CC, 1, 0,
                 P[TR_B3] + l * CC, sB, CC, 0, 1.f, 1.f, 0, II, CC, CC, 1);
        }
        ln_kernel<<<(II + 7) / 8, 256, 0, stream>>>(sB, P[TR_LN_G], P[TR_LN_B], sB, II, CC);

        // backbone update + rigid compose
        backbone_kernel<<<(II + 255) / 256, 256, 0, stream>>>(
            sB, P[BB_WB], P[BB_BB], P[BB_WC], P[BB_BC], P[BB_WD], P[BB_BD],
            P[BB_WT], P[BB_BT], RB, tB);
    }

    // ---- angle resnet ----
    // a = relu(s)@Win + bin + relu(s0)@Winit + binit
    gemm(stream, sB, CC, 0, 1, P[ANG_WIN], CC, 0, 0, angA, CC, 1, 0,
         P[ANG_BIN], nullptr, 0, 0, 0.f, 1.f, 0, II, CC, CC, 1);
    gemm(stream, s0, CC, 0, 1, P[ANG_WINIT], CC, 0, 0, angA, CC, 1, 0,
         P[ANG_BINIT], angA, CC, 0, 1.f, 1.f, 0, II, CC, CC, 1);
    for (int l = 0; l < NLL; ++l) {
        const float* W1 = P[ANG_W1] + (long)l * CC * CC;
        const float* W2 = P[ANG_W2] + (long)l * CC * CC;
        // t = relu( relu(a)@W1 + b1 )
        gemm(stream, angA, CC, 0, 1, W1, CC, 0, 0, angT, CC, 1, 0,
             P[ANG_B1] + l * CC, nullptr, 0, 0, 0.f, 1.f, 1, II, CC, CC, 1);
        // a = 2*a + t@W2 + b2
        gemm(stream, angT, CC, 0, 0, W2, CC, 0, 0, angA, CC, 1, 0,
             P[ANG_B2] + l * CC, angA, CC, 0, 2.f, 1.f, 0, II, CC, CC, 1);
    }
    // a14 = relu(a) @ Wout + bout
    gemm(stream, angA, CC, 0, 1, P[ANG_WOUT], 14, 0, 0, a14, 14, 1, 0,
         P[ANG_BOUT], nullptr, 0, 0, 0.f, 1.f, 0, II, 14, CC, 1);

    // ---- write outputs: [s (49152) | R (3456) | t (1152) | unnorm (5376) | norm (5376)] ----
    float* out = (float*)d_out;
    copy_kernel<<<((long)II * CC + 255) / 256, 256, 0, stream>>>(sB, out, (long)II * CC);
    copy_kernel<<<((long)II * 9 + 255) / 256, 256, 0, stream>>>(RB, out + II * CC, (long)II * 9);
    copy_kernel<<<((long)II * 3 + 255) / 256, 256, 0, stream>>>(tB, out + II * CC + II * 9, (long)II * 3);
    angle_final_kernel<<<(II * 7 + 255) / 256, 256, 0, stream>>>(
        a14, out + II * CC + II * 9 + II * 3, out + II * CC + II * 9 + II * 3 + II * 14);
}
